// KnowledgeGNN_81853486727884
// MI455X (gfx1250) — matrix operand order repaired and checked by reading, not compiled, not run on hardware
//
#include <hip/hip_runtime.h>

namespace {

constexpr int D   = 64;
constexpr int T   = 4096;
constexpr int N   = 20000;
constexpr int M   = T + N;      // 24096
constexpr int E   = 100000;
constexpr int R   = 50;
constexpr int TOK = 768;

typedef __attribute__((ext_vector_type(16))) __bf16 v16bf;
typedef __attribute__((ext_vector_type(8)))  float  v8f;

// ---------------------------------------------------------------- utilities
__global__ void kgnn_zero_f32(float* __restrict__ p, int n) {
  int i = blockIdx.x * blockDim.x + threadIdx.x;
  if (i < n) p[i] = 0.f;
}

// x[T+n, :] = kg_emb[node_ids[n], :]
__global__ void kgnn_gather_nodes(const int* __restrict__ node_ids,
                                  const float* __restrict__ kg_emb,
                                  float* __restrict__ x) {
  int i = blockIdx.x * blockDim.x + threadIdx.x;     // over N*D
  if (i >= N * D) return;
  int n = i >> 6, d = i & 63;
  x[(size_t)(T + n) * D + d] = kg_emb[(size_t)node_ids[n] * D + d];
}

// relation id = argmax of one-hot row (exact for one-hot edge_attr)
__global__ void kgnn_compute_rel(const float* __restrict__ edge_attr,
                                 int* __restrict__ rel) {
  int e = blockIdx.x * blockDim.x + threadIdx.x;
  if (e >= E) return;
  const float* row = edge_attr + (size_t)e * R;
  int best = 0; float bv = row[0];
  for (int c = 1; c < R; ++c) { float v = row[c]; if (v > bv) { bv = v; best = c; } }
  rel[e] = best;
}

// ------------------------------------------------- WMMA bf16 GEMM (fp32 acc)
// C[Mg x Ng] = A[Mg x KG] @ B[KG x Ng] (+bias[Ng]) (+add[Mg x Ng]) (relu)
// KG compile-time multiple of 32; Ng multiple of 64.
// Block = 4 waves sharing one 64-wide N tile; each wave owns a 16(M)x64(N)
// macro tile. B tile is cooperatively staged through LDS, pre-swizzled into
// the WMMA B-fragment layout so each lane reads its fragment as contiguous
// 32-byte LDS accesses. All 4 B fragments are loaded before the WMMA burst
// so ds returns overlap with matrix-pipe issue (partial dscnt waits).
template<int KG, bool BIAS, bool ADD, bool RELU, bool MGUARD>
__global__ __launch_bounds__(128)
void kgnn_gemm_wmma(const float* __restrict__ A, const float* __restrict__ B,
                    const float* __restrict__ bias, const float* __restrict__ add,
                    float* __restrict__ C, int Mg, int Ng) {
  // [nTile][laneSlot][8 x packed bf16 pair] = 4KB
  __shared__ unsigned int BsU[4][32][8];

  int lane = threadIdx.x;            // 0..31
  int wv   = threadIdx.y;            // 0..3
  int tid  = wv * 32 + lane;         // 0..127
  int tileM = (blockIdx.y * 4 + wv) * 16;
  int tileN = blockIdx.x * 64;
  bool active = tileM < Mg;          // inactive waves still run barriers

  int half = lane >> 4, l15 = lane & 15;

  // A row owned by this lane (clamped in-bounds; MGUARD zeroes padded rows)
  int m = tileM + l15;
  float rowScale = 1.f;
  if (MGUARD) rowScale = (m < Mg) ? 1.f : 0.f;
  if (m > Mg - 1) m = Mg - 1;
  const float* Arow = A + (size_t)m * KG;

  // staging coords: thread stages column c of the 64-wide tile,
  // K rows g*16 + {0,2,...,14}(+1)  ->  fragment st, lane slot sl
  int c  = tid & 63;
  int g  = tid >> 6;                 // 0..1
  int st = c >> 4;
  int sl = 16 * g + (c & 15);
  const float* Bcol = B + tileN + c;

  v8f acc[4] = {{}, {}, {}, {}};

  for (int k0 = 0; k0 < KG; k0 += 32) {
    // ---- cooperative, coalesced stage of B[k0..k0+32) x 64 cols into LDS
    #pragma unroll
    for (int p = 0; p < 8; ++p) {
      int k = k0 + g * 16 + 2 * p;
      float b0 = Bcol[(size_t)k * Ng];
      float b1 = Bcol[(size_t)(k + 1) * Ng];
      union { unsigned int u; __bf16 h[2]; } pk;
      pk.h[0] = (__bf16)b0;
      pk.h[1] = (__bf16)b1;
      BsU[st][sl][p] = pk.u;         // one ds_store_b32 per K-pair
    }
    __syncthreads();

    // ---- A fragment: 16-bit A 16x32 layout, contiguous float2 K-pairs
    v16bf a;
    #pragma unroll
    for (int j = 0; j < 8; ++j) {
      int kb = k0 + 16 * half + 2 * j;
      float2 av = *(const float2*)(Arow + kb);
      a[2 * j]     = (__bf16)(av.x * rowScale);
      a[2 * j + 1] = (__bf16)(av.y * rowScale);
    }

    // ---- issue all 4 B-fragment loads, then burst 4 WMMAs (one A reuse)
    v16bf bfr[4];
    #pragma unroll
    for (int t = 0; t < 4; ++t)
      bfr[t] = *(const v16bf*)(&BsU[t][lane][0]);
    #pragma unroll
    for (int t = 0; t < 4; ++t)
      acc[t] = __builtin_amdgcn_wmma_f32_16x16x32_bf16(false, a, false, bfr[t],
                                                       (short)0, acc[t], false, false);
    __syncthreads();                 // protect LDS before next stage
  }

  // D layout: VGPR r -> row (r + 8*half), column l15
  if (active) {
    #pragma unroll
    for (int t = 0; t < 4; ++t) {
      int n = tileN + t * 16 + l15;
      #pragma unroll
      for (int r = 0; r < 8; ++r) {
        int row = tileM + r + 8 * half;
        if (!MGUARD || row < Mg) {
          float v = acc[t][r];
          if (BIAS) v += bias[n];
          if (ADD)  v += add[(size_t)row * Ng + n];
          if (RELU) v = fmaxf(v, 0.f);
          C[(size_t)row * Ng + n] = v;
        }
      }
    }
  }
}

// --------------------------------------------- per-edge attention logits
// logit[e] = x_i . att_w[0:64] + (x_j @ W_rel[rel]) . att_w[64:128] + att_b
__global__ void kgnn_edge_logits(const float* __restrict__ xin, const int* __restrict__ src,
                                 const int* __restrict__ dst, const int* __restrict__ rel,
                                 const float* __restrict__ wrel, const float* __restrict__ att_w,
                                 const float* __restrict__ att_b, float* __restrict__ logits) {
  int e = blockIdx.x;
  int d = threadIdx.x;                       // 0..63 (output dim o == d)
  __shared__ float xj[D];
  __shared__ float sm[D];
  int s = src[e], t = dst[e], r = rel[e];
  xj[d] = xin[(size_t)s * D + d];
  __syncthreads();
  const float* W = wrel + (size_t)r * D * D; // row-major [d_in][d_out]
  float nbr = 0.f;
  #pragma unroll 8
  for (int k = 0; k < D; ++k) nbr = fmaf(xj[k], W[k * D + d], nbr);
  sm[d] = xin[(size_t)t * D + d] * att_w[d] + nbr * att_w[D + d];
  __syncthreads();
  for (int sft = D / 2; sft > 0; sft >>= 1) {
    if (d < sft) sm[d] += sm[d + sft];
    __syncthreads();
  }
  if (d == 0) logits[e] = sm[0] + att_b[0];
}

// global softmax stats over all E logits (single block, two passes)
__global__ void kgnn_softmax_reduce(const float* __restrict__ logits, float* __restrict__ red) {
  __shared__ float sm[1024];
  int tid = threadIdx.x;
  float mx = -3.4e38f;
  for (int i = tid; i < E; i += 1024) mx = fmaxf(mx, logits[i]);
  sm[tid] = mx; __syncthreads();
  for (int s = 512; s > 0; s >>= 1) { if (tid < s) sm[tid] = fmaxf(sm[tid], sm[tid + s]); __syncthreads(); }
  float gmax = sm[0]; __syncthreads();
  float acc = 0.f;
  for (int i = tid; i < E; i += 1024) acc += __expf(logits[i] - gmax);
  sm[tid] = acc; __syncthreads();
  for (int s = 512; s > 0; s >>= 1) { if (tid < s) sm[tid] += sm[tid + s]; __syncthreads(); }
  if (tid == 0) { red[0] = gmax; red[1] = sm[0]; }
}

// aggr[dst] += x_j * softmax(logit)
__global__ void kgnn_scatter_msgs(const float* __restrict__ xin, const int* __restrict__ src,
                                  const int* __restrict__ dst, const float* __restrict__ logits,
                                  const float* __restrict__ red, float* __restrict__ aggr) {
  long long i = (long long)blockIdx.x * blockDim.x + threadIdx.x;
  if (i >= (long long)E * D) return;
  int e = (int)(i >> 6), d = (int)(i & 63);
  float alpha = __expf(logits[e] - red[0]) / red[1];
  atomicAdd(&aggr[(size_t)dst[e] * D + d], xin[(size_t)src[e] * D + d] * alpha);
}

// masked MSE over [E, D]; mask = relation not among last 3
__global__ void kgnn_kge_loss(const float* __restrict__ fx, const int* __restrict__ src,
                              const int* __restrict__ dst, const int* __restrict__ rel,
                              const float* __restrict__ rel_emb, float* __restrict__ acc) {
  int e = blockIdx.x, d = threadIdx.x;
  __shared__ float sm[D];
  int r = rel[e];
  float v = 0.f;
  if (r < R - 3) {
    float diff = fx[(size_t)src[e] * D + d] + rel_emb[(size_t)r * D + d]
               - fx[(size_t)dst[e] * D + d];
    v = diff * diff;
  }
  sm[d] = v; __syncthreads();
  for (int s = D / 2; s > 0; s >>= 1) { if (d < s) sm[d] += sm[d + s]; __syncthreads(); }
  if (d == 0) atomicAdd(acc, sm[0]);
}

// node-type cross entropy (3 classes)
__global__ void kgnn_nt_loss(const float* __restrict__ fx, const int* __restrict__ labels,
                             const float* __restrict__ w_nt, const float* __restrict__ b_nt,
                             float* __restrict__ acc) {
  int m = blockIdx.x * blockDim.x + threadIdx.x;
  if (m >= M) return;
  float l0 = b_nt[0], l1 = b_nt[1], l2 = b_nt[2];
  const float* row = fx + (size_t)m * D;
  #pragma unroll 8
  for (int d = 0; d < D; ++d) {
    float v = row[d];
    l0 = fmaf(v, w_nt[d * 3 + 0], l0);
    l1 = fmaf(v, w_nt[d * 3 + 1], l1);
    l2 = fmaf(v, w_nt[d * 3 + 2], l2);
  }
  float mx  = fmaxf(l0, fmaxf(l1, l2));
  float lse = mx + __logf(__expf(l0 - mx) + __expf(l1 - mx) + __expf(l2 - mx));
  int lb = labels[m];
  float sel = (lb == 0) ? l0 : ((lb == 1) ? l1 : l2);
  atomicAdd(acc, lse - sel);
}

__global__ void kgnn_finalize(const float* __restrict__ red, float* __restrict__ out) {
  out[0] = red[2] / (float)((long long)E * D);   // kge_loss
  out[1] = red[3] / (float)M;                    // nt_loss
}

} // namespace

extern "C" void kernel_launch(void* const* d_in, const int* in_sizes, int n_in,
                              void* d_out, int out_size, void* d_ws, size_t ws_size,
                              hipStream_t stream) {
  (void)in_sizes; (void)n_in; (void)out_size; (void)ws_size;
  const int*   node_ids         = (const int*)  d_in[0];
  const int*   edge_index       = (const int*)  d_in[1];
  const float* edge_attr        = (const float*)d_in[2];
  const float* token_embeddings = (const float*)d_in[3];
  const int*   node_type_labels = (const int*)  d_in[4];
  const float* kg_emb  = (const float*)d_in[5];
  const float* rel_emb = (const float*)d_in[6];
  const float* w_e2    = (const float*)d_in[7];
  const float* b_e2    = (const float*)d_in[8];
  const float* att1_w  = (const float*)d_in[9];
  const float* att1_b  = (const float*)d_in[10];
  const float* root1   = (const float*)d_in[11];
  const float* bias1   = (const float*)d_in[12];
  const float* att2_w  = (const float*)d_in[13];
  const float* att2_b  = (const float*)d_in[14];
  const float* root2   = (const float*)d_in[15];
  const float* bias2   = (const float*)d_in[16];
  const float* w_lin1  = (const float*)d_in[17];
  const float* b_lin1  = (const float*)d_in[18];
  const float* w_lin2  = (const float*)d_in[19];
  const float* b_lin2  = (const float*)d_in[20];
  const float* w_nt    = (const float*)d_in[21];
  const float* b_nt    = (const float*)d_in[22];

  const int* src = edge_index;
  const int* dst = edge_index + E;

  // workspace carve-up (floats)
  float* ws = (float*)d_ws;
  size_t off = 0;
  float* x      = ws + off; off += (size_t)M * D;      // layer input
  float* h      = ws + off; off += (size_t)M * D;      // relu(layer1)
  float* fx     = ws + off; off += (size_t)M * D;      // final_x
  float* aggr   = ws + off; off += (size_t)M * D;      // scatter accumulator
  float* wrel   = ws + off; off += (size_t)R * D * D;  // 50 distinct edge matrices (800KB, L2-resident)
  float* logits = ws + off; off += (size_t)E;
  float* red    = ws + off; off += 8;                  // [max, sum, kge_acc, nt_acc]
  int*   rel    = (int*)(ws + off);                    // E ints

  float* out = (float*)d_out;

  dim3 gblk(32, 4);
  auto mgrid = [](int Mg, int Ng) {
    return dim3((unsigned)(Ng / 64), (unsigned)(((Mg + 15) / 16 + 3) / 4));
  };

  kgnn_zero_f32<<<1, 32, 0, stream>>>(red, 8);
  kgnn_gather_nodes<<<(N * D + 255) / 256, 256, 0, stream>>>(node_ids, kg_emb, x);

  // tok = tokens @ w_lin1 + b_lin1   [4096 x 768] x [768 x 64]
  kgnn_gemm_wmma<TOK, true, false, false, false>
      <<<mgrid(T, D), gblk, 0, stream>>>(token_embeddings, w_lin1, b_lin1, nullptr, x, T, D);

  kgnn_compute_rel<<<(E + 255) / 256, 256, 0, stream>>>(edge_attr, rel);

  // per-relation edge matrices: relu(rel_emb @ w_e2 + b_e2)  [50 x 64] x [64 x 4096]
  kgnn_gemm_wmma<D, true, false, true, true>
      <<<mgrid(R, D * D), gblk, 0, stream>>>(rel_emb, w_e2, b_e2, nullptr, wrel, R, D * D);

  // ---- conv layer 1 ----
  kgnn_zero_f32<<<(M * D + 255) / 256, 256, 0, stream>>>(aggr, M * D);
  kgnn_edge_logits<<<E, D, 0, stream>>>(x, src, dst, rel, wrel, att1_w, att1_b, logits);
  kgnn_softmax_reduce<<<1, 1024, 0, stream>>>(logits, red);
  kgnn_scatter_msgs<<<(E * D + 255) / 256, 256, 0, stream>>>(x, src, dst, logits, red, aggr);
  // h = relu(aggr + x @ root1 + bias1)
  kgnn_gemm_wmma<D, true, true, true, false>
      <<<mgrid(M, D), gblk, 0, stream>>>(x, root1, bias1, aggr, h, M, D);

  // ---- conv layer 2 ----
  kgnn_zero_f32<<<(M * D + 255) / 256, 256, 0, stream>>>(aggr, M * D);
  kgnn_edge_logits<<<E, D, 0, stream>>>(h, src, dst, rel, wrel, att2_w, att2_b, logits);
  kgnn_softmax_reduce<<<1, 1024, 0, stream>>>(logits, red);
  kgnn_scatter_msgs<<<(E * D + 255) / 256, 256, 0, stream>>>(h, src, dst, logits, red, aggr);
  // final_x = aggr + h @ root2 + bias2
  kgnn_gemm_wmma<D, true, true, false, false>
      <<<mgrid(M, D), gblk, 0, stream>>>(h, root2, bias2, aggr, fx, M, D);

  // final_outputs = final_x @ w_lin2 + b_lin2   [24096 x 64] x [64 x 768]
  kgnn_gemm_wmma<D, true, false, false, false>
      <<<mgrid(M, TOK), gblk, 0, stream>>>(fx, w_lin2, b_lin2, nullptr, out, M, TOK);

  // losses
  kgnn_kge_loss<<<E, D, 0, stream>>>(fx, src, dst, rel, rel_emb, red + 2);
  kgnn_nt_loss<<<(M + 255) / 256, 256, 0, stream>>>(fx, node_type_labels, w_nt, b_nt, red + 3);
  kgnn_finalize<<<1, 1, 0, stream>>>(red, out + (size_t)M * TOK);
}
